// MetaPathRRWPEncoder_39539468927440
// MI455X (gfx1250) — matrix operand order repaired and checked by reading, not compiled, not run on hardware
//
#include <hip/hip_runtime.h>
#include <hip/hip_bf16.h>

// ---------------------------------------------------------------------------
// MetaPath RRWP encoder for MI455X (gfx1250).
// 14 x (2048^3) GEMMs dominate -> v_wmma_f32_16x16x32_bf16 (f32 accumulate),
// bf16 storage in HBM, async global->LDS staging, double-buffered LDS.
// ---------------------------------------------------------------------------

typedef __attribute__((ext_vector_type(16))) __bf16 v16bf;
typedef __attribute__((ext_vector_type(8)))  __bf16 v8bf;
typedef __attribute__((ext_vector_type(8)))  float  v8f;
typedef __attribute__((ext_vector_type(4)))  float  v4f;
typedef __attribute__((ext_vector_type(4)))  unsigned int v4u;
typedef int v4i __attribute__((vector_size(16)));  // matches async-builtin pointee

#define NDIM 2048
#define NN   ((size_t)NDIM * (size_t)NDIM)
#define BM 128
#define BN 128
#define BK 32
#define NPAN (NDIM / BK)

#if __has_builtin(__builtin_amdgcn_global_load_async_to_lds_b128)
#define HAVE_ASYNC 1
#endif

#ifdef HAVE_ASYNC
#if __has_builtin(__builtin_amdgcn_s_wait_asynccnt)
#define WAIT_ASYNC(n) __builtin_amdgcn_s_wait_asynccnt(n)
#else
#define WAIT_ASYNC(n) asm volatile("s_wait_asynccnt %0" ::"i"(n) : "memory")
#endif
#else
#define WAIT_ASYNC(n) ((void)0)
#endif

// 16-byte global -> LDS copy (async DMA path on gfx1250 when available).
static __device__ __forceinline__ void copy16(const __bf16* g, __bf16* l) {
#ifdef HAVE_ASYNC
  __builtin_amdgcn_global_load_async_to_lds_b128(
      (__attribute__((address_space(1))) v4i*)g,
      (__attribute__((address_space(3))) v4i*)l, 0, 0);
#else
  *(v4u*)l = *(const v4u*)g;
#endif
}

static __device__ __forceinline__ v16bf cat8(v8bf lo, v8bf hi) {
  return __builtin_shufflevector(lo, hi, 0, 1, 2, 3, 4, 5, 6, 7,
                                         8, 9, 10, 11, 12, 13, 14, 15);
}

// ---------------------------------------------------------------------------
__global__ void zero_f32(float* __restrict__ p, int n) {
  int i = blockIdx.x * blockDim.x + threadIdx.x;
  if (i < n) p[i] = 0.0f;
}

// adj[t][src][dst] += 1  (duplicate edges sum, matching sparse mm semantics)
__global__ void scatter_adj(const int* __restrict__ edge_index,
                            const int* __restrict__ edge_type,
                            float* __restrict__ adj, int nedges) {
  int i = blockIdx.x * blockDim.x + threadIdx.x;
  if (i >= nedges) return;
  int s = edge_index[i];
  int d = edge_index[nedges + i];
  int t = edge_type[i];
  atomicAdd(adj + (size_t)t * NN + (size_t)s * NDIM + d, 1.0f);
}

// fp32 plane -> bf16 plane, optionally transposed (to build B^T operands).
__global__ void f32_to_bf16(const float* __restrict__ src,
                            __bf16* __restrict__ dst, int transpose) {
  int i = blockIdx.x * blockDim.x + threadIdx.x;
  if (i >= (int)NN) return;
  int r = i >> 11, c = i & (NDIM - 1);
  __bf16 v = (__bf16)src[i];
  dst[transpose ? ((size_t)c << 11) + r : (size_t)i] = v;
}

// ---------------------------------------------------------------------------
// C = A @ B with B given transposed (BT row-major = B column-major), bf16 in,
// f32 WMMA accumulate, bf16 out. Optionally also writes CT = C^T (so C can be
// consumed as a B-operand by later GEMMs with pure-copy staging).
// Block tile 128x128, K-panel 32, 8 waves (2Mx4N), wave tile 64x32.
// Double-buffered LDS fed by GLOBAL_LOAD_ASYNC_TO_LDS_B128.
// ---------------------------------------------------------------------------
__global__ __launch_bounds__(256)
void gemm_wmma_bf16(const __bf16* __restrict__ A, const __bf16* __restrict__ BT,
                    __bf16* __restrict__ C, __bf16* __restrict__ CT) {
  __shared__ __align__(16) __bf16 lA[2][BM][BK];
  __shared__ __align__(16) __bf16 lB[2][BN][BK];

  const int tid  = threadIdx.x;
  const int lane = tid & 31;
  const int wave = tid >> 5;   // 0..7
  const int wm   = wave >> 2;  // 0..1 -> rows wm*64
  const int wn   = wave & 3;   // 0..3 -> cols wn*32
  const int half = lane >> 4;  // half-wave select
  const int lr   = lane & 15;

  const size_t bm = (size_t)blockIdx.y * BM;
  const size_t bn = (size_t)blockIdx.x * BN;

  v8f acc[4][2] = {};

  // Stage one 128x32 bf16 panel of A and of BT: 512 16B chunks each,
  // 2 chunks per thread per matrix -> 4 async copies per thread.
  auto stage = [&](int pan, int buf) {
    const int k0 = pan * BK;
#pragma unroll
    for (int s = 0; s < 2; ++s) {
      int c = tid + s * 256;
      int r = c >> 2, off = (c & 3) * 8;
      copy16(A  + (bm + r) * NDIM + k0 + off, &lA[buf][r][off]);
      copy16(BT + (bn + r) * NDIM + k0 + off, &lB[buf][r][off]);
    }
  };

  stage(0, 0);

  for (int p = 0; p < NPAN; ++p) {
    const int cur = p & 1;
    if (p + 1 < NPAN) {
      stage(p + 1, cur ^ 1);   // stream next panel into the other buffer
      WAIT_ASYNC(4);           // current panel's 4 copies have completed
    } else {
      WAIT_ASYNC(0);
    }
    if (p + 2 < NPAN && tid < BM)  // keep L2 warm two panels ahead
      __builtin_prefetch(&A[(bm + tid) * NDIM + (size_t)(p + 2) * BK], 0, 1);
    __syncthreads();

    // A fragments: lane<16 holds M=lr, K={0..7,16..23}; lane>=16 K+=8.
    v16bf afrag[4];
#pragma unroll
    for (int mt = 0; mt < 4; ++mt) {
      const __bf16* row = &lA[cur][wm * 64 + mt * 16 + lr][0];
      afrag[mt] = cat8(*(const v8bf*)(row + 8 * half),
                       *(const v8bf*)(row + 16 + 8 * half));
    }
    // B fragments: lane<16 holds N=lr, K={0..7,16..23}; lane>=16 K+=8.
    v16bf bfrag[2];
#pragma unroll
    for (int nt = 0; nt < 2; ++nt) {
      const __bf16* col = &lB[cur][wn * 32 + nt * 16 + lr][0];
      bfrag[nt] = cat8(*(const v8bf*)(col + 8 * half),
                       *(const v8bf*)(col + 16 + 8 * half));
    }

#pragma unroll
    for (int mt = 0; mt < 4; ++mt)
#pragma unroll
      for (int nt = 0; nt < 2; ++nt)
        acc[mt][nt] = __builtin_amdgcn_wmma_f32_16x16x32_bf16(
            false, afrag[mt], false, bfrag[nt],
            (short)0, acc[mt][nt], false, false);

    __syncthreads();
  }

  // Epilogue: D layout - VGPR r: lanes<16 M=r, lanes>=16 M=r+8; N=lane&15.
#pragma unroll
  for (int mt = 0; mt < 4; ++mt) {
#pragma unroll
    for (int nt = 0; nt < 2; ++nt) {
      size_t col = bn + wn * 32 + nt * 16 + lr;
#pragma unroll
      for (int r = 0; r < 8; ++r) {
        size_t row = bm + wm * 64 + mt * 16 + 8 * half + r;
        __bf16 v = (__bf16)acc[mt][nt][r];
        C[row * NDIM + col] = v;
        if (CT) CT[col * NDIM + row] = v;
      }
    }
  }
}

// ---------------------------------------------------------------------------
// out[n][p][m][e] = b_p[e] + (n==m)*W_p[0][e] + sum_{k=1..7} T_p^k[n,m]*W_p[k][e]
// pw holds 14 bf16 planes: plane(p,k) = p*7 + (k-1).
// ---------------------------------------------------------------------------
__global__ __launch_bounds__(256)
void rrwp_emit(const __bf16* __restrict__ pw,
               const float* __restrict__ W0, const float* __restrict__ b0,
               const float* __restrict__ W1, const float* __restrict__ b1,
               float* __restrict__ out) {
  size_t gid = (size_t)blockIdx.x * blockDim.x + threadIdx.x;
  if (gid >= 2 * NN) return;
  int m = (int)(gid % NDIM);
  size_t t = gid / NDIM;
  int p = (int)(t & 1);
  int n = (int)(t >> 1);

  const float* W = p ? W1 : W0;  // [8][8] row-major
  const float* b = p ? b1 : b0;

  float accv[8];
#pragma unroll
  for (int e = 0; e < 8; ++e)
    accv[e] = b[e] + (n == m ? W[e] : 0.0f);

  const __bf16* base = pw + (size_t)p * 7 * NN + (size_t)n * NDIM + m;
#pragma unroll
  for (int k = 1; k <= 7; ++k) {
    float v = (float)__builtin_nontemporal_load(base + (size_t)(k - 1) * NN);
#pragma unroll
    for (int e = 0; e < 8; ++e)
      accv[e] += v * W[k * 8 + e];
  }

  v4f lo = {accv[0], accv[1], accv[2], accv[3]};
  v4f hi = {accv[4], accv[5], accv[6], accv[7]};
  __builtin_nontemporal_store(lo, (v4f*)(out + gid * 8));
  __builtin_nontemporal_store(hi, (v4f*)(out + gid * 8 + 4));
}

// ---------------------------------------------------------------------------
// ws layout, bf16 planes of NN elements (8 MB each), 20 planes = 160 MB:
//   0..6   : T0^1..T0^7      7..13 : T1^1..T1^7
//   14,15  : T0^T, T1^T      16,17 : adj1^T, adj3^T     18,19 : adj0, adj2
// fp32 scatter scratch (4 planes, 64 MB) overlaps bf16 planes 0..7, which are
// only written by GEMMs after the scratch is dead.
// ---------------------------------------------------------------------------
extern "C" void kernel_launch(void* const* d_in, const int* in_sizes, int n_in,
                              void* d_out, int out_size, void* d_ws, size_t ws_size,
                              hipStream_t stream) {
  const int*   edge_index = (const int*)d_in[0];  // [2, E]
  const int*   edge_type  = (const int*)d_in[1];  // [E]
  const float* W0 = (const float*)d_in[3];
  const float* b0 = (const float*)d_in[4];
  const float* W1 = (const float*)d_in[5];
  const float* b1 = (const float*)d_in[6];
  const int nedges = in_sizes[1];

  __bf16* planes = (__bf16*)d_ws;
  auto P = [&](int i) { return planes + (size_t)i * NN; };
  float* adjf = (float*)d_ws;  // overlaps planes 0..7 (dead before GEMMs)

  // 1) fp32 adjacency scatter
  int zn = (int)(4 * NN);
  zero_f32<<<(zn + 255) / 256, 256, 0, stream>>>(adjf, zn);
  scatter_adj<<<(nedges + 255) / 256, 256, 0, stream>>>(edge_index, edge_type,
                                                        adjf, nedges);

  // 2) convert to bf16; B-side operands transposed for copy-only staging
  int cg = (int)((NN + 255) / 256);
  f32_to_bf16<<<cg, 256, 0, stream>>>(adjf + 0 * NN, P(18), 0);  // adj0
  f32_to_bf16<<<cg, 256, 0, stream>>>(adjf + 1 * NN, P(16), 1);  // adj1^T
  f32_to_bf16<<<cg, 256, 0, stream>>>(adjf + 2 * NN, P(19), 0);  // adj2
  f32_to_bf16<<<cg, 256, 0, stream>>>(adjf + 3 * NN, P(17), 1);  // adj3^T

  // 3) T_p = adj(2p) @ adj(2p+1); also emit T_p^T for the power chain
  dim3 gg(NDIM / BN, NDIM / BM);
  gemm_wmma_bf16<<<gg, 256, 0, stream>>>(P(18), P(16), P(0), P(14));
  gemm_wmma_bf16<<<gg, 256, 0, stream>>>(P(19), P(17), P(7), P(15));

  // 4) power chains: T_p^k = T_p^{k-1} @ T_p
  for (int k = 2; k <= 7; ++k) {
    gemm_wmma_bf16<<<gg, 256, 0, stream>>>(P(k - 2), P(14), P(k - 1), nullptr);
    gemm_wmma_bf16<<<gg, 256, 0, stream>>>(P(7 + k - 2), P(15), P(7 + k - 1),
                                           nullptr);
  }

  // 5) fused linear + bias + identity, nontemporal 256 MB store
  size_t nthreads = 2 * NN;
  rrwp_emit<<<(unsigned)((nthreads + 255) / 256), 256, 0, stream>>>(
      planes, W0, b0, W1, b1, (float*)d_out);
}
